// EncoderDecoder_87995289960500
// MI455X (gfx1250) — matrix operand order repaired and checked by reading, not compiled, not run on hardware
//
#include <hip/hip_runtime.h>
#include <hip/hip_bf16.h>
#include <math.h>

// ---------------------------------------------------------------------------
// Problem constants (from reference)
// ---------------------------------------------------------------------------
#define Bsz   4096
#define Tt    365
#define Ff    64          // input features
#define SBc   8
#define FBc   16
#define Nh    128         // hidden
#define G4    512         // 4*Nh gate width
#define OUTS  30
#define ENCS  16          // encoder sequence length (1 seasonal + 7 days + 8 bb rows)

#define BT      16              // batch rows per workgroup tile
#define NT      (Bsz / BT)      // 256 batch tiles
#define NTILES  (G4 / 16)       // 32 N-tiles of 16 columns
#define KBX     (Ff / 32)       // 2  k-blocks for x (K=64)
#define KBH     (Nh / 32)       // 4  k-blocks for h (K=128)

typedef __attribute__((ext_vector_type(16))) __bf16 v16bf;
typedef __attribute__((ext_vector_type(8)))  float  v8f;

// A-fragment K index for 16-bit 16x32 A matrix (ISA 7.12.2):
// lane 0-15 / 16-31 -> M = lane%16 ; element e of v16bf -> K below
__device__ __forceinline__ int a_k(int lane, int e) {
    return ((e >> 3) << 4) + ((lane >> 4) << 3) + (e & 7);
}

// Fast gate nonlinearities: v_exp_f32 + v_rcp_f32 instead of IEEE divide chain.
__device__ __forceinline__ float sigm(float x) {
    return __builtin_amdgcn_rcpf(1.0f + __expf(-x));
}
__device__ __forceinline__ float tanh_fast(float x) {
    // tanh(x) = 2*sigmoid(2x) - 1
    return __builtin_fmaf(2.0f, __builtin_amdgcn_rcpf(1.0f + __expf(-2.0f * x)), -1.0f);
}

// ---------------------------------------------------------------------------
// Prep 1: build encoder input sequence as bf16 A-fragments.
// x_seq[b, s, :]: s=0 -> a[b, T-14, :]; s=1..7 -> a[b, T-8+s, :];
//                 s=8..15 -> relu(b[b, s-8, :] @ W_combine + b_combine)
// Layout: xf[((tile*16 + s)*KBX + kblk)*32 + lane][16 bf16]
// ---------------------------------------------------------------------------
__global__ void build_xfrag(const float* __restrict__ a, const float* __restrict__ b,
                            const float* __restrict__ Wc, const float* __restrict__ bc,
                            __bf16* __restrict__ xf) {
    int tile = blockIdx.x >> 4;
    int s    = blockIdx.x & 15;
    int kblk = threadIdx.x >> 5;
    int lane = threadIdx.x & 31;
    int m     = lane & 15;
    int batch = tile * BT + m;
    __bf16* dst = xf + ((size_t)(((tile * ENCS + s) * KBX + kblk) * 32 + lane) << 4);
    for (int e = 0; e < 16; ++e) {
        int k = kblk * 32 + a_k(lane, e);
        float v;
        if (s < 8) {
            int t = (s == 0) ? (Tt - 14) : (Tt - 8 + s);
            v = a[(size_t)batch * Tt * Ff + (size_t)t * Ff + k];
        } else {
            int j = s - 8;
            float acc = bc[k];
            const float* brow = b + (size_t)batch * SBc * FBc + j * FBc;
            for (int q = 0; q < FBc; ++q) acc += brow[q] * Wc[q * Ff + k];
            v = fmaxf(acc, 0.0f);
        }
        dst[e] = (__bf16)v;
    }
}

// ---------------------------------------------------------------------------
// Prep 2: weight (Kdim x 512) -> bf16 B-fragments.
// B-fragment (32x16 bf16): lane -> N = lane%16 ; element e -> K = a_k(lane, e)
// Layout: wf[(nblk*(Kdim/32) + kblk)*32 + lane][16 bf16]
// ---------------------------------------------------------------------------
__global__ void build_wfrag(const float* __restrict__ W, __bf16* __restrict__ wf, int Kdim) {
    int kbc  = Kdim >> 5;
    int nblk = blockIdx.x / kbc;
    int kblk = blockIdx.x % kbc;
    int lane = threadIdx.x;
    int n    = nblk * 16 + (lane & 15);
    __bf16* dst = wf + ((size_t)((nblk * kbc + kblk) * 32 + lane) << 4);
    for (int e = 0; e < 16; ++e) {
        int k = kblk * 32 + a_k(lane, e);
        dst[e] = (__bf16)W[(size_t)k * G4 + n];
    }
}

// ---------------------------------------------------------------------------
// Prep 3: x0 = relu(a[:, T-1, :] @ W_first + b_first) -> bf16 A-fragments
// Layout: x0f[(tile*KBH + kblk)*32 + lane][16 bf16]
// ---------------------------------------------------------------------------
__global__ void build_x0frag(const float* __restrict__ a, const float* __restrict__ Wf,
                             const float* __restrict__ bfst, __bf16* __restrict__ x0f) {
    int tile = blockIdx.x >> 2;
    int kblk = blockIdx.x & 3;
    int lane = threadIdx.x;
    int m     = lane & 15;
    int batch = tile * BT + m;
    const float* arow = a + (size_t)batch * Tt * Ff + (size_t)(Tt - 1) * Ff;
    __bf16* dst = x0f + ((size_t)((tile * KBH + kblk) * 32 + lane) << 4);
    for (int e = 0; e < 16; ++e) {
        int k = kblk * 32 + a_k(lane, e);
        float acc = bfst[k];
        for (int q = 0; q < Ff; ++q) acc += arow[q] * Wf[q * Nh + k];
        dst[e] = (__bf16)fmaxf(acc, 0.0f);
    }
}

// ---------------------------------------------------------------------------
// Encoder: 16 recurrent steps. Per WG: 16 batch rows, 8 waves x 4 N-tiles.
// Weight B-fragments end up register-resident across the whole loop (seen in
// codegen); A-fragments stream from global (x) and LDS (h).
// ---------------------------------------------------------------------------
__global__ __launch_bounds__(256)
void lstm_encoder(const __bf16* __restrict__ xf, const __bf16* __restrict__ wkf,
                  const __bf16* __restrict__ wrf, const float* __restrict__ bias,
                  __bf16* __restrict__ hfg, float* __restrict__ cg) {
    __shared__ float zbuf[BT][G4 + 4];
    __shared__ float cbuf[BT * Nh];
    __shared__ __align__(32) __bf16 hfrag[KBH * 32 * 16];
    __shared__ float biasS[G4];

    int tid  = threadIdx.x;
    int lane = tid & 31;
    int wave = tid >> 5;
    int tile = blockIdx.x;

    for (int i = tid; i < BT * Nh; i += 256) cbuf[i] = 0.0f;
    for (int i = tid; i < KBH * 32 * 16; i += 256) hfrag[i] = (__bf16)0.0f;
    for (int i = tid; i < G4; i += 256) biasS[i] = bias[i];
    __syncthreads();

    for (int s = 0; s < ENCS; ++s) {
        // Hoist A-fragments (shared by this wave's 4 N-tiles)
        const __bf16* xbase = xf + ((size_t)((tile * ENCS + s) * KBX) * 32 << 4);
        v16bf ax[KBX];
        #pragma unroll
        for (int kb = 0; kb < KBX; ++kb)
            ax[kb] = *reinterpret_cast<const v16bf*>(xbase + ((kb * 32 + lane) << 4));
        v16bf ah[KBH];
        #pragma unroll
        for (int kb = 0; kb < KBH; ++kb)
            ah[kb] = *reinterpret_cast<const v16bf*>(&hfrag[(kb * 32 + lane) << 4]);

        // Prefetch next step's x fragments into L2/L0 while we compute.
        if (s + 1 < ENCS) {
            const __bf16* xnext = xf + ((size_t)((tile * ENCS + s + 1) * KBX) * 32 << 4);
            __builtin_prefetch(xnext + (lane << 4), 0, 3);
            __builtin_prefetch(xnext + ((32 + lane) << 4), 0, 3);
        }

        for (int j = 0; j < 4; ++j) {
            int nb = wave * 4 + j;
            v8f acc = {};
            #pragma unroll
            for (int kb = 0; kb < KBX; ++kb) {
                v16bf bf = *reinterpret_cast<const v16bf*>(
                    wkf + ((size_t)((nb * KBX + kb) * 32 + lane) << 4));
                acc = __builtin_amdgcn_wmma_f32_16x16x32_bf16(
                    false, ax[kb], false, bf, (short)0, acc, false, false);
            }
            #pragma unroll
            for (int kb = 0; kb < KBH; ++kb) {
                v16bf bf = *reinterpret_cast<const v16bf*>(
                    wrf + ((size_t)((nb * KBH + kb) * 32 + lane) << 4));
                acc = __builtin_amdgcn_wmma_f32_16x16x32_bf16(
                    false, ah[kb], false, bf, (short)0, acc, false, false);
            }
            // D layout: lane L, reg r -> M = (L/16)*8 + r, N = L%16
            #pragma unroll
            for (int r = 0; r < 8; ++r) {
                int mrow = ((lane >> 4) << 3) + r;
                zbuf[mrow][nb * 16 + (lane & 15)] = acc[r];
            }
        }
        __syncthreads();

        // Fused gate update: c = sig(f)*c + sig(i)*tanh(g); h = sig(o)*tanh(c)
        for (int u = 0; u < (BT * Nh) / 256; ++u) {
            int idx = u * 256 + tid;
            int m = idx >> 7, k = idx & 127;
            float zi = zbuf[m][k]          + biasS[k];
            float zfv= zbuf[m][Nh + k]     + biasS[Nh + k];
            float zg = zbuf[m][2 * Nh + k] + biasS[2 * Nh + k];
            float zo = zbuf[m][3 * Nh + k] + biasS[3 * Nh + k];
            float c  = sigm(zfv) * cbuf[idx] + sigm(zi) * tanh_fast(zg);
            float h  = sigm(zo) * tanh_fast(c);
            cbuf[idx] = c;
            // write h back in A-fragment order
            int kb = k >> 5, r = k & 31;
            int ln = m + (((r >> 3) & 1) << 4);
            int e  = ((r >> 4) << 3) + (r & 7);
            hfrag[((kb * 32 + ln) << 4) + e] = (__bf16)h;
        }
        __syncthreads();
    }
    for (int i = tid; i < KBH * 32 * 16; i += 256)
        hfg[(size_t)tile * (KBH * 32 * 16) + i] = hfrag[i];
    for (int i = tid; i < BT * Nh; i += 256)
        cg[(size_t)tile * BT * Nh + i] = cbuf[i];
}

// ---------------------------------------------------------------------------
// Decoder: 30 steps, xin fed back with h. Fuses preds @ W_out per step.
// ---------------------------------------------------------------------------
__global__ __launch_bounds__(256)
void lstm_decoder(const __bf16* __restrict__ x0f, const __bf16* __restrict__ hfg,
                  const float* __restrict__ cg, const __bf16* __restrict__ wkf,
                  const __bf16* __restrict__ wrf, const float* __restrict__ bias,
                  const float* __restrict__ Wout, const float* __restrict__ bout,
                  float* __restrict__ out) {
    __shared__ float zbuf[BT][G4 + 4];
    __shared__ float cbuf[BT * Nh];
    __shared__ float hlin[BT * Nh];
    __shared__ __align__(32) __bf16 hfrag[KBH * 32 * 16];
    __shared__ __align__(32) __bf16 xinfrag[KBH * 32 * 16];
    __shared__ float biasS[G4];
    __shared__ float woutS[Nh];

    int tid  = threadIdx.x;
    int lane = tid & 31;
    int wave = tid >> 5;
    int tile = blockIdx.x;

    for (int i = tid; i < BT * Nh; i += 256) cbuf[i] = cg[(size_t)tile * BT * Nh + i];
    for (int i = tid; i < KBH * 32 * 16; i += 256) {
        hfrag[i]   = hfg[(size_t)tile * (KBH * 32 * 16) + i];
        xinfrag[i] = x0f[(size_t)tile * (KBH * 32 * 16) + i];
    }
    for (int i = tid; i < G4; i += 256) biasS[i] = bias[i];
    for (int i = tid; i < Nh; i += 256) woutS[i] = Wout[i];
    __syncthreads();

    float b_out0 = bout[0];

    for (int t = 0; t < OUTS; ++t) {
        v16bf axin[KBH], ah[KBH];
        #pragma unroll
        for (int kb = 0; kb < KBH; ++kb) {
            axin[kb] = *reinterpret_cast<const v16bf*>(&xinfrag[(kb * 32 + lane) << 4]);
            ah[kb]   = *reinterpret_cast<const v16bf*>(&hfrag[(kb * 32 + lane) << 4]);
        }
        for (int j = 0; j < 4; ++j) {
            int nb = wave * 4 + j;
            v8f acc = {};
            #pragma unroll
            for (int kb = 0; kb < KBH; ++kb) {
                v16bf bf = *reinterpret_cast<const v16bf*>(
                    wkf + ((size_t)((nb * KBH + kb) * 32 + lane) << 4));
                acc = __builtin_amdgcn_wmma_f32_16x16x32_bf16(
                    false, axin[kb], false, bf, (short)0, acc, false, false);
            }
            #pragma unroll
            for (int kb = 0; kb < KBH; ++kb) {
                v16bf bf = *reinterpret_cast<const v16bf*>(
                    wrf + ((size_t)((nb * KBH + kb) * 32 + lane) << 4));
                acc = __builtin_amdgcn_wmma_f32_16x16x32_bf16(
                    false, ah[kb], false, bf, (short)0, acc, false, false);
            }
            #pragma unroll
            for (int r = 0; r < 8; ++r) {
                int mrow = ((lane >> 4) << 3) + r;
                zbuf[mrow][nb * 16 + (lane & 15)] = acc[r];
            }
        }
        __syncthreads();

        for (int u = 0; u < (BT * Nh) / 256; ++u) {
            int idx = u * 256 + tid;
            int m = idx >> 7, k = idx & 127;
            float zi = zbuf[m][k]          + biasS[k];
            float zfv= zbuf[m][Nh + k]     + biasS[Nh + k];
            float zg = zbuf[m][2 * Nh + k] + biasS[2 * Nh + k];
            float zo = zbuf[m][3 * Nh + k] + biasS[3 * Nh + k];
            float c  = sigm(zfv) * cbuf[idx] + sigm(zi) * tanh_fast(zg);
            float h  = sigm(zo) * tanh_fast(c);
            cbuf[idx] = c;
            hlin[idx] = h;
            int kb = k >> 5, r = k & 31;
            int ln = m + (((r >> 3) & 1) << 4);
            int e  = ((r >> 4) << 3) + (r & 7);
            __bf16 hb = (__bf16)h;
            hfrag[((kb * 32 + ln) << 4) + e]   = hb;  // next-step recurrent input
            xinfrag[((kb * 32 + ln) << 4) + e] = hb;  // next-step feed-forward input
        }
        __syncthreads();

        if (tid < BT) {
            float s = b_out0;
            for (int k = 0; k < Nh; ++k) s += hlin[tid * Nh + k] * woutS[k];
            out[(size_t)(tile * BT + tid) * OUTS + t] = s;
        }
        __syncthreads();
    }
}

// ---------------------------------------------------------------------------
// Launch
// ---------------------------------------------------------------------------
extern "C" void kernel_launch(void* const* d_in, const int* in_sizes, int n_in,
                              void* d_out, int out_size, void* d_ws, size_t ws_size,
                              hipStream_t stream) {
    (void)in_sizes; (void)n_in; (void)out_size; (void)ws_size;
    const float* a        = (const float*)d_in[0];
    const float* b        = (const float*)d_in[1];
    const float* Wc       = (const float*)d_in[2];
    const float* bc       = (const float*)d_in[3];
    const float* W_first  = (const float*)d_in[4];
    const float* b_first  = (const float*)d_in[5];
    const float* Wk_enc   = (const float*)d_in[6];
    const float* Wr_enc   = (const float*)d_in[7];
    const float* bias_enc = (const float*)d_in[8];
    const float* Wk_dec   = (const float*)d_in[9];
    const float* Wr_dec   = (const float*)d_in[10];
    const float* bias_dec = (const float*)d_in[11];
    const float* W_out    = (const float*)d_in[12];
    const float* b_out    = (const float*)d_in[13];
    float* out = (float*)d_out;

    char* ws = (char*)d_ws;
    size_t off = 0;
    __bf16* xfrag = (__bf16*)(ws + off); off += (size_t)NT * ENCS * KBX * 512 * 2; // 8 MB
    __bf16* x0f   = (__bf16*)(ws + off); off += (size_t)NT * KBH * 512 * 2;        // 1 MB
    __bf16* hfg   = (__bf16*)(ws + off); off += (size_t)NT * KBH * 512 * 2;        // 1 MB
    float*  cg    = (float*)(ws + off);  off += (size_t)Bsz * Nh * 4;              // 2 MB
    __bf16* wkenc = (__bf16*)(ws + off); off += (size_t)NTILES * KBX * 512 * 2;    // 64 KB
    __bf16* wrenc = (__bf16*)(ws + off); off += (size_t)NTILES * KBH * 512 * 2;    // 128 KB
    __bf16* wkdec = (__bf16*)(ws + off); off += (size_t)NTILES * KBH * 512 * 2;    // 128 KB
    __bf16* wrdec = (__bf16*)(ws + off); off += (size_t)NTILES * KBH * 512 * 2;    // 128 KB

    build_xfrag<<<NT * ENCS, 64, 0, stream>>>(a, b, Wc, bc, xfrag);
    build_wfrag<<<NTILES * KBX, 32, 0, stream>>>(Wk_enc, wkenc, Ff);
    build_wfrag<<<NTILES * KBH, 32, 0, stream>>>(Wr_enc, wrenc, Nh);
    build_wfrag<<<NTILES * KBH, 32, 0, stream>>>(Wk_dec, wkdec, Nh);
    build_wfrag<<<NTILES * KBH, 32, 0, stream>>>(Wr_dec, wrdec, Nh);
    build_x0frag<<<NT * KBH, 32, 0, stream>>>(a, W_first, b_first, x0f);

    lstm_encoder<<<NT, 256, 0, stream>>>(xfrag, wkenc, wrenc, bias_enc, hfg, cg);
    lstm_decoder<<<NT, 256, 0, stream>>>(x0f, hfg, cg, wkdec, wrdec, bias_dec,
                                         W_out, b_out, out);
}
